// WaveNetModel_69020124446872
// MI455X (gfx1250) — matrix operand order repaired and checked by reading, not compiled, not run on hardware
//
#include <hip/hip_runtime.h>
#include <math.h>

typedef __attribute__((ext_vector_type(16))) __bf16 v16bf;
typedef __attribute__((ext_vector_type(8)))  __bf16 bf16x8;
typedef __attribute__((ext_vector_type(4)))  __bf16 bf16x4;
typedef __attribute__((ext_vector_type(8)))  float  v8f;
typedef __attribute__((ext_vector_type(4)))  float  f32x4;

#define BATCH   8
#define TLEN    16384
#define CH      32
#define NBLK    6
#define NCLS    2
#define TILE_T  64
#define HALO    32
#define ROWS    (TILE_T + 2 * HALO)   // 128

__device__ __forceinline__ __bf16 tobf(float f) { return (__bf16)f; }

// Branch-free transcendentals: gfx1250 has native v_tanh_f32 / v_exp_f32.
__device__ __forceinline__ float fast_tanh(float x) {
#if __has_builtin(__builtin_amdgcn_tanhf)
    return __builtin_amdgcn_tanhf(x);
#else
    float ax = fabsf(x);
    float e  = __expf(2.f * ax);
    float r  = 1.f - 2.f * __builtin_amdgcn_rcpf(e + 1.f);
    return copysignf(r, x);
#endif
}
__device__ __forceinline__ float fast_sigmoid(float x) {
    return __builtin_amdgcn_rcpf(1.f + __expf(-x));
}

// Wave-local LDS RAW fence: LDS ops from one wave complete in order, so
// waiting DScnt==0 (plus a compile-time memory barrier) orders a wave's
// own store->load through LDS without an s_barrier.
__device__ __forceinline__ void wave_lds_fence() {
    asm volatile("s_wait_dscnt 0x0" ::: "memory");
}

union frag_u { v16bf v; bf16x8 h[2]; };

// A fragment (16 time x 32 cin, bf16) from an LDS tile with 64B rows.
// A layout: lanes 0-15: M=lane, K = 0..15;
//           lanes 16-31: M=lane-16, K = 16..31.
// Each lane's data = two contiguous 16B chunks -> 2x ds_load_b128.
__device__ __forceinline__ v16bf frag_A(const __bf16* tile, int row_base, int lane) {
    int m  = lane & 15;
    int c0 = (lane >> 4) & 1;                       // chunk 0 or 1
    const bf16x8* rp = (const bf16x8*)(tile + (row_base + m) * CH);
    frag_u u;
    u.h[0] = rp[2 * c0];
    u.h[1] = rp[2 * c0 + 1];
    return u.v;
}

// B fragment from TRANSPOSED LDS weights wT[cout][cin] (64B rows).
// Lanes 0-15: N=lane, K=0..15; lanes 16-31: N=lane-16, K=16..31.
// Each lane's 16 K-values are contiguous -> 2x ds_load_b128.
__device__ __forceinline__ v16bf frag_B(const __bf16* wT, int nbase, int lane) {
    int n  = nbase + (lane & 15);
    int k0 = (lane >> 4) << 4;                      // 0 or 16
    const bf16x8* rp = (const bf16x8*)(wT + n * CH + k0);
    frag_u u;
    u.h[0] = rp[0];
    u.h[1] = rp[1];
    return u.v;
}

// C init = bias broadcast: C[M,N] = bias[N]; N = nbase + lane%16 for all 8 rows.
__device__ __forceinline__ v8f bias_C(const float* bias, int nbase, int lane) {
    float v = bias[nbase + (lane & 15)];
    v8f c;
#pragma unroll
    for (int r = 0; r < 8; ++r) c[r] = v;
    return c;
}

__global__ void __launch_bounds__(256) k_init(
    const float* __restrict__ x, const float* __restrict__ w_init,
    const float* __restrict__ b_init, float* __restrict__ h,
    float* __restrict__ skip_sum, float* __restrict__ logits) {
    int bt = blockIdx.x * blockDim.x + threadIdx.x;   // over B*T
    if (blockIdx.x == 0 && threadIdx.x < BATCH * NCLS) logits[threadIdx.x] = 0.f;
    if (bt >= BATCH * TLEN) return;
    float xv = x[bt];
    f32x4* hp = (f32x4*)(h + (size_t)bt * CH);
    f32x4* sp = (f32x4*)(skip_sum + (size_t)bt * CH);
#pragma unroll
    for (int q = 0; q < CH / 4; ++q) {
        f32x4 v, z;
#pragma unroll
        for (int e = 0; e < 4; ++e) {
            int c = q * 4 + e;
            v[e] = fmaxf(xv * w_init[c] + b_init[c], 0.f);
            z[e] = 0.f;
        }
        hp[q] = v;
        sp[q] = z;
    }
}

__global__ void __launch_bounds__(128) k_block(
    const float* __restrict__ h_in, float* __restrict__ h_out,
    float* __restrict__ skip_sum,
    const float* __restrict__ w_tanh, const float* __restrict__ b_tanh,
    const float* __restrict__ w_sig,  const float* __restrict__ b_sig,
    const float* __restrict__ w_skip, const float* __restrict__ b_skip,
    int dil) {
    __shared__ alignas(16) __bf16 sh_h[ROWS * CH];      // 8 KB tile + halo
    __shared__ alignas(16) __bf16 sh_wt[3 * CH * CH];   // 6 KB tanh W^T
    __shared__ alignas(16) __bf16 sh_wg[3 * CH * CH];   // 6 KB sig  W^T
    __shared__ alignas(16) __bf16 sh_wk[CH * CH];       // 2 KB skip W^T
    __shared__ alignas(16) __bf16 sh_u[4 * 16 * CH];    // 4 KB gated product
    __shared__ alignas(16) float  sh_s[4 * 16 * CH];    // 8 KB f32 skip tile

    const int tid  = threadIdx.x;
    const int lane = tid & 31;
    const int wave = tid >> 5;
    const int tiles_per_b = TLEN / TILE_T;
    const int b  = blockIdx.x / tiles_per_b;
    const int t0 = (blockIdx.x % tiles_per_b) * TILE_T;

    // Stage weights TRANSPOSED: global [k][cin][cout] -> LDS [k][cout][cin].
    for (int i = tid; i < 3 * CH * CH; i += 128) {
        int k = i >> 10, ci = (i >> 5) & 31, co = i & 31;
        int tdst = (k << 10) + (co << 5) + ci;
        sh_wt[tdst] = tobf(w_tanh[i]);
        sh_wg[tdst] = tobf(w_sig[i]);
        if (k == 0) sh_wk[(co << 5) + ci] = tobf(w_skip[i]);
    }
    // Stage h tile + halo: float4 global loads, bf16x4 LDS stores.
    for (int i = tid; i < ROWS * (CH / 4); i += 128) {
        int r = i / (CH / 4), q = i % (CH / 4);
        int t = t0 - HALO + r;
        f32x4 v = {0.f, 0.f, 0.f, 0.f};
        if (t >= 0 && t < TLEN)
            v = *(const f32x4*)(h_in + (((size_t)b * TLEN + t) * CH) + q * 4);
        bf16x4 o;
#pragma unroll
        for (int e = 0; e < 4; ++e) o[e] = tobf(v[e]);
        *(bf16x4*)(sh_h + r * CH + q * 4) = o;
    }
    __syncthreads();   // cross-wave: tile + weights visible to all waves

    const int rbase = HALO + wave * 16;

    v8f ct0 = bias_C(b_tanh, 0, lane), ct1 = bias_C(b_tanh, 16, lane);
    v8f cg0 = bias_C(b_sig,  0, lane), cg1 = bias_C(b_sig, 16, lane);

#pragma unroll
    for (int k = 0; k < 3; ++k) {
        int off = (k - 1) * dil;
        v16bf a = frag_A(sh_h, rbase + off, lane);
        v16bf bt0 = frag_B(sh_wt + k * CH * CH, 0, lane);
        ct0 = __builtin_amdgcn_wmma_f32_16x16x32_bf16(false, a, false, bt0, (short)0, ct0, false, false);
        v16bf bt1 = frag_B(sh_wt + k * CH * CH, 16, lane);
        ct1 = __builtin_amdgcn_wmma_f32_16x16x32_bf16(false, a, false, bt1, (short)0, ct1, false, false);
        v16bf bg0 = frag_B(sh_wg + k * CH * CH, 0, lane);
        cg0 = __builtin_amdgcn_wmma_f32_16x16x32_bf16(false, a, false, bg0, (short)0, cg0, false, false);
        v16bf bg1 = frag_B(sh_wg + k * CH * CH, 16, lane);
        cg1 = __builtin_amdgcn_wmma_f32_16x16x32_bf16(false, a, false, bg1, (short)0, cg1, false, false);
    }

    // gated activation: u = tanh(conv_t) * sigmoid(conv_g), branch-free
#pragma unroll
    for (int r = 0; r < 8; ++r) {
        ct0[r] = fast_tanh(ct0[r]) * fast_sigmoid(cg0[r]);
        ct1[r] = fast_tanh(ct1[r]) * fast_sigmoid(cg1[r]);
    }

    // stash u row-major [16 x 32] in this wave's own LDS region (C-layout scatter)
    __bf16* su = sh_u + wave * 16 * CH;
    {
        int m0 = (lane >> 4) << 3;   // 0 or 8
        int n  = lane & 15;
#pragma unroll
        for (int r = 0; r < 8; ++r) {
            su[(m0 + r) * CH + n]      = tobf(ct0[r]);
            su[(m0 + r) * CH + 16 + n] = tobf(ct1[r]);
        }
    }
    wave_lds_fence();   // wave-local RAW through LDS; no s_barrier needed

    // skip = relu(1x1 conv of u)
    v16bf au  = frag_A(su, 0, lane);
    v8f  ck0  = bias_C(b_skip, 0, lane);
    v16bf bk0 = frag_B(sh_wk, 0, lane);
    ck0 = __builtin_amdgcn_wmma_f32_16x16x32_bf16(false, au, false, bk0, (short)0, ck0, false, false);
    v8f  ck1  = bias_C(b_skip, 16, lane);
    v16bf bk1 = frag_B(sh_wk, 16, lane);
    ck1 = __builtin_amdgcn_wmma_f32_16x16x32_bf16(false, au, false, bk1, (short)0, ck1, false, false);

    // scatter f32 skip tile to this wave's LDS region (C-layout)
    {
        float* ss = sh_s + wave * 16 * CH;
        int m0 = (lane >> 4) << 3;
        int n  = lane & 15;
#pragma unroll
        for (int r = 0; r < 8; ++r) {
            ss[(m0 + r) * CH + n]      = fmaxf(ck0[r], 0.f);
            ss[(m0 + r) * CH + 16 + n] = fmaxf(ck1[r], 0.f);
        }
    }
    wave_lds_fence();   // wave-local RAW through LDS

    // residual + skip accumulation, fully coalesced b128 traffic
    {
        size_t gbase = ((size_t)b * TLEN + t0 + wave * 16) * CH;   // 512 floats
        const f32x4* hin4  = (const f32x4*)(h_in + gbase);
        f32x4*       hout4 = (f32x4*)(h_out + gbase);
        f32x4*       ssg4  = (f32x4*)(skip_sum + gbase);
        const f32x4* sl4   = (const f32x4*)(sh_s + wave * 16 * CH);
#pragma unroll
        for (int j = 0; j < 4; ++j) {
            int idx = j * 32 + lane;
            f32x4 sk = sl4[idx];
            f32x4 hv = hin4[idx];
            f32x4 sv = ssg4[idx];
            f32x4 ho, so;
#pragma unroll
            for (int e = 0; e < 4; ++e) { ho[e] = hv[e] + sk[e]; so[e] = sv[e] + sk[e]; }
            hout4[idx] = ho;
            ssg4[idx]  = so;
        }
    }
}

__global__ void __launch_bounds__(256) k_final(
    const float* __restrict__ skip_sum,
    const float* __restrict__ w_final, const float* __restrict__ b_final,
    const float* __restrict__ w_dense, float* __restrict__ logits) {
    __shared__ float swf[CH * CH];
    __shared__ float sred[256][2];
    int tid = threadIdx.x;
    for (int i = tid; i < CH * CH; i += 256) swf[i] = w_final[i];
    __syncthreads();
    int blocks_per_b = TLEN / 256;
    int b = blockIdx.x / blocks_per_b;
    int t = (blockIdx.x % blocks_per_b) * 256 + tid;
    const f32x4* sp4 = (const f32x4*)(skip_sum + ((size_t)b * TLEN + t) * CH);
    float s[CH];
#pragma unroll
    for (int q = 0; q < CH / 4; ++q) {
        f32x4 v = sp4[q];
#pragma unroll
        for (int e = 0; e < 4; ++e) s[q * 4 + e] = fmaxf(v[e], 0.f);
    }
    float p0 = 0.f, p1 = 0.f;
    const float* wd = w_dense + (size_t)t * CH * NCLS;
#pragma unroll 4
    for (int co = 0; co < CH; ++co) {
        float acc = b_final[co];
#pragma unroll
        for (int ci = 0; ci < CH; ++ci) acc += s[ci] * swf[ci * CH + co];
        float z = fmaxf(acc, 0.f);
        p0 += z * wd[co * NCLS + 0];
        p1 += z * wd[co * NCLS + 1];
    }
    sred[tid][0] = p0; sred[tid][1] = p1;
    __syncthreads();
    for (int s2 = 128; s2 > 0; s2 >>= 1) {
        if (tid < s2) { sred[tid][0] += sred[tid + s2][0]; sred[tid][1] += sred[tid + s2][1]; }
        __syncthreads();
    }
    if (tid == 0) {
        atomicAdd(&logits[b * NCLS + 0], sred[0][0]);
        atomicAdd(&logits[b * NCLS + 1], sred[0][1]);
    }
}

__global__ void k_softmax(const float* __restrict__ logits,
                          const float* __restrict__ b_dense,
                          float* __restrict__ out) {
    int b = threadIdx.x;
    if (b < BATCH) {
        float l0 = logits[b * NCLS + 0] + b_dense[0];
        float l1 = logits[b * NCLS + 1] + b_dense[1];
        float m  = fmaxf(l0, l1);
        float e0 = __expf(l0 - m), e1 = __expf(l1 - m);
        float inv = __builtin_amdgcn_rcpf(e0 + e1);
        out[b * NCLS + 0] = e0 * inv;
        out[b * NCLS + 1] = e1 * inv;
    }
}

extern "C" void kernel_launch(void* const* d_in, const int* in_sizes, int n_in,
                              void* d_out, int out_size, void* d_ws, size_t ws_size,
                              hipStream_t stream) {
    (void)in_sizes; (void)n_in; (void)out_size; (void)ws_size;
    const float* x       = (const float*)d_in[0];
    const float* w_init  = (const float*)d_in[1];
    const float* b_init  = (const float*)d_in[2];
    const float* w_tanh  = (const float*)d_in[3];
    const float* b_tanh  = (const float*)d_in[4];
    const float* w_sig   = (const float*)d_in[5];
    const float* b_sig   = (const float*)d_in[6];
    const float* w_skip  = (const float*)d_in[7];
    const float* b_skip  = (const float*)d_in[8];
    const float* w_final = (const float*)d_in[9];
    const float* b_final = (const float*)d_in[10];
    const float* w_dense = (const float*)d_in[11];
    const float* b_dense = (const float*)d_in[12];
    float* out = (float*)d_out;

    const size_t N = (size_t)BATCH * TLEN * CH;   // 4,194,304 floats
    float* ws        = (float*)d_ws;
    float* h0        = ws;
    float* h1        = ws + N;
    float* skip_sum  = ws + 2 * N;
    float* logits    = ws + 3 * N;                // 16 floats

    k_init<<<(BATCH * TLEN + 255) / 256, 256, 0, stream>>>(x, w_init, b_init, h0, skip_sum, logits);

    float* hin = h0; float* hout = h1;
    for (int i = 0; i < NBLK; ++i) {
        k_block<<<BATCH * (TLEN / TILE_T), 128, 0, stream>>>(
            hin, hout, skip_sum,
            w_tanh + (size_t)i * 3 * CH * CH, b_tanh + (size_t)i * CH,
            w_sig  + (size_t)i * 3 * CH * CH, b_sig  + (size_t)i * CH,
            w_skip + (size_t)i * CH * CH,     b_skip + (size_t)i * CH,
            1 << i);
        float* tmp = hin; hin = hout; hout = tmp;
    }

    k_final<<<BATCH * (TLEN / 256), 256, 0, stream>>>(skip_sum, w_final, b_final, w_dense, logits);
    k_softmax<<<1, 32, 0, stream>>>(logits, b_dense, out);
}
